// MoELayer_47184510714167
// MI455X (gfx1250) — matrix operand order repaired and checked by reading, not compiled, not run on hardware
//
#include <hip/hip_runtime.h>

// MoE all-experts FFN for MI455X (gfx1250, wave32) -- COMPUTE-BOUND design.
//
// FLOPs: 2 * 8192*1024*4096 * 2 layers * 8 experts = 1.1 TFLOP.
// At dense-bf16 WMMA peak (~2.5-5 PFLOPS) that is ~220-440us, vs a 14us HBM
// floor (320MB @ 23.3TB/s) -> compute-bound by >20x. Therefore:
//  * convert fp32 weights/activations to bf16 ONCE (prep kernels), never in
//    the GEMM hot loop;
//  * weights in WMMA panel layout [K/32][N][32]: a B fragment is two
//    contiguous global_load_b128 per lane (compiler clause-pairs them);
//  * unfused two-GEMM pipeline per expert; gate folded into H rows
//    (g*(H@W2) == (g(*)H)@W2) so GEMM2 is plain accumulation; the 64MB
//    per-expert H buffer is produced-then-consumed -> L2-resident (192MB L2);
//  * A tiles double-buffered in LDS via CDNA5 async copies
//    (global_load_async_to_lds_b128 + s_wait_asynccnt). K-chunk = 64 so each
//    wait+barrier covers 16 WMMAs (halved sync overhead vs K-chunk 32).
//
// Workspace requirement: ~209MB (gate 256KB + xb 16MB + W1b/W2b 64MB each +
// Hb 64MB, reused across experts).

#define Bsz   8192
#define DIN   1024
#define DHID  4096
#define DOUT  1024
#define NEXP  8

typedef __attribute__((ext_vector_type(16))) __bf16       v16bf;
typedef __attribute__((ext_vector_type(4)))  __bf16       v4bf;
typedef __attribute__((ext_vector_type(8)))  float        v8f;
typedef __attribute__((ext_vector_type(4)))  float        f32x4;
typedef __attribute__((ext_vector_type(4)))  unsigned int u32x4;

__device__ __forceinline__ v8f zero8() {
  v8f z = {0.f, 0.f, 0.f, 0.f, 0.f, 0.f, 0.f, 0.f};
  return z;
}

__device__ __forceinline__ v8f wmma_bf16(v16bf a, v16bf b, v8f c) {
  return __builtin_amdgcn_wmma_f32_16x16x32_bf16(false, a, false, b, (short)0, c,
                                                 false, false);
}

// A fragment (16x32 bf16) from a [rows][64] bf16 chunk in LDS (ld = 64).
// ISA 16-bit A layout: lanes 0-15 row M=lane, K {0..7,16..23}; lanes 16-31
// row M=lane-16, K {8..15,24..31}. Two ds_load_b128 per lane.
__device__ __forceinline__ v16bf load_a_lds64(const unsigned short* S, int m0,
                                              int ksub, int lane) {
  const unsigned short* p =
      S + (m0 + (lane & 15)) * 64 + ksub * 32 + ((lane >> 4) << 3);
  union { u32x4 q[2]; v16bf v; } u;
  u.q[0] = *(const u32x4*)(p);
  u.q[1] = *(const u32x4*)(p + 16);
  return u.v;
}

// B fragment (32x16 bf16) from panel layout [K/32][N][32]: column n's 32
// K-values are contiguous -> two global_load_b128 per lane (K {0..7,16..23} / {8..15,24..31}).
__device__ __forceinline__ v16bf load_b_panel(const unsigned short* __restrict__ P,
                                              int N, int k0, int n, int lane) {
  const unsigned short* p =
      P + ((size_t)(k0 >> 5) * N + n + (lane & 15)) * 32 + ((lane >> 4) << 3);
  union { u32x4 q[2]; v16bf v; } u;
  u.q[0] = *(const u32x4*)(p);
  u.q[1] = *(const u32x4*)(p + 16);
  return u.v;
}

// Async-copy one 128x64 bf16 A chunk (16KB) into an LDS buffer.
// 1024 x 16B segments, 256 threads x 4. Tracked by ASYNCcnt.
__device__ __forceinline__ void stage_async(const unsigned short* __restrict__ Agl,
                                            size_t ldA, int k, unsigned ldsOff,
                                            int tid) {
#pragma unroll
  for (int s = 0; s < 4; ++s) {
    const int seg = tid * 4 + s;     // 0..1023
    const int row = seg >> 3;        // 8 x 16B segments per 128B row
    const int si  = seg & 7;
    const unsigned long long g =
        (unsigned long long)(Agl + (size_t)row * ldA + k + si * 8);
    const unsigned l = ldsOff + (unsigned)(row * 128 + si * 16);
    asm volatile("global_load_async_to_lds_b128 %0, %1, off"
                 :: "v"(l), "v"(g) : "memory");
  }
}

__device__ __forceinline__ void wait_async_all() {
  asm volatile("s_wait_asynccnt 0x0" ::: "memory");
}

// ---------------------------------------------------------------------------
// Prep: x fp32 -> bf16 row-major.
// ---------------------------------------------------------------------------
__global__ void __launch_bounds__(256)
cvt_x_kernel(const float* __restrict__ x, __bf16* __restrict__ xb) {
  const size_t i = ((size_t)blockIdx.x * 256 + threadIdx.x) * 4;
  f32x4 v = *(const f32x4*)(x + i);
  v4bf o = { (__bf16)v.x, (__bf16)v.y, (__bf16)v.z, (__bf16)v.w };
  *(v4bf*)(xb + i) = o;
}

// ---------------------------------------------------------------------------
// Prep: weights fp32 [E][K][N] -> bf16 panel layout [E][K/32][N][32].
// Coalesced reads along N; element (k,n) -> ((k>>5)*N + n)*32 + (k&31).
// ---------------------------------------------------------------------------
template <int KT, int NN>
__global__ void __launch_bounds__(256)
cvt_w_kernel(const float* __restrict__ W, __bf16* __restrict__ Wb) {
  const size_t base = ((size_t)blockIdx.x * 256 + threadIdx.x) * 4;  // flat e,k,n
  const int    n    = (int)(base % NN);
  const size_t ek   = base / NN;
  const int    k    = (int)(ek % KT);
  const size_t e    = ek / KT;
  f32x4 v = *(const f32x4*)(W + base);
  __bf16* P = Wb + e * (size_t)KT * NN + ((size_t)(k >> 5) * NN) * 32 + (k & 31);
  P[(size_t)(n + 0) * 32] = (__bf16)v.x;
  P[(size_t)(n + 1) * 32] = (__bf16)v.y;
  P[(size_t)(n + 2) * 32] = (__bf16)v.z;
  P[(size_t)(n + 3) * 32] = (__bf16)v.w;
}

// ---------------------------------------------------------------------------
// Gate: one wave per row, gate[b,:] = softmax(x[b] @ Wg + bg).
// ---------------------------------------------------------------------------
__global__ void __launch_bounds__(256)
moe_gate(const float* __restrict__ x, const float* __restrict__ Wg,
         const float* __restrict__ bg, float* __restrict__ gate) {
  const int lane = threadIdx.x & 31;
  const int wave = threadIdx.x >> 5;
  const int b    = blockIdx.x * 8 + wave;

  float acc[NEXP];
#pragma unroll
  for (int e = 0; e < NEXP; ++e) acc[e] = 0.f;

  const float* xr = x + (size_t)b * DIN;
  for (int k = lane; k < DIN; k += 32) {
    const float xv = xr[k];
    const float* wr = Wg + (size_t)k * NEXP;
#pragma unroll
    for (int e = 0; e < NEXP; ++e) acc[e] += xv * wr[e];
  }
#pragma unroll
  for (int off = 16; off > 0; off >>= 1) {
#pragma unroll
    for (int e = 0; e < NEXP; ++e) acc[e] += __shfl_down(acc[e], off, 32);
  }
  if (lane == 0) {
    float mx = -3.0e38f;
#pragma unroll
    for (int e = 0; e < NEXP; ++e) { acc[e] += bg[e]; mx = acc[e] > mx ? acc[e] : mx; }
    float s = 0.f;
#pragma unroll
    for (int e = 0; e < NEXP; ++e) { acc[e] = __expf(acc[e] - mx); s += acc[e]; }
    const float inv = 1.f / s;
#pragma unroll
    for (int e = 0; e < NEXP; ++e) gate[(size_t)b * NEXP + e] = acc[e] * inv;
  }
}

// ---------------------------------------------------------------------------
// GEMM1 (per expert): Hb[m,n] = gate[m,e] * relu(xb[m,:] @ W1b[e][:,n] + b1)
// Block 256 thr = 8 waves; block tile 128x128; wave tile 32x64 (2x4 frags).
// ---------------------------------------------------------------------------
__global__ void __launch_bounds__(256)
gemm1_kernel(const unsigned short* __restrict__ xb,
             const unsigned short* __restrict__ W1b,
             const float* __restrict__ b1, const float* __restrict__ gate,
             __bf16* __restrict__ Hb, int e) {
  __shared__ __align__(16) unsigned short Asm[2][128 * 64];

  const int tid  = threadIdx.x;
  const int lane = tid & 31;
  const int wave = tid >> 5;
  const int m0   = blockIdx.x * 128;
  const int n0   = blockIdx.y * 128;
  const int m0w  = (wave & 3) * 32;
  const int n0w  = (wave >> 2) * 64;

  const unsigned short* Agl = xb + (size_t)m0 * DIN;
  const unsigned short* Bp  = W1b + (size_t)e * DIN * DHID;  // panel layout
  const float* b1e = b1 + (size_t)e * DHID;

  const unsigned lds0 = (unsigned)(unsigned long long)(&Asm[0][0]);
  const unsigned lds1 = (unsigned)(unsigned long long)(&Asm[1][0]);

  v8f acc[2][4];
#pragma unroll
  for (int i = 0; i < 2; ++i)
#pragma unroll
    for (int j = 0; j < 4; ++j) acc[i][j] = zero8();

  const int NS = DIN / 64;
  stage_async(Agl, DIN, 0, lds0, tid);
  for (int ks = 0; ks < NS; ++ks) {
    wait_async_all();
    __syncthreads();
    if (ks + 1 < NS)
      stage_async(Agl, DIN, (ks + 1) * 64, ((ks + 1) & 1) ? lds1 : lds0, tid);

    const unsigned short* S = &Asm[ks & 1][0];
#pragma unroll
    for (int ksub = 0; ksub < 2; ++ksub) {
      const v16bf a0 = load_a_lds64(S, m0w,      ksub, lane);
      const v16bf a1 = load_a_lds64(S, m0w + 16, ksub, lane);
      const int k0 = ks * 64 + ksub * 32;
#pragma unroll
      for (int j = 0; j < 4; ++j) {
        const v16bf bf = load_b_panel(Bp, DHID, k0, n0 + n0w + j * 16, lane);
        acc[0][j] = wmma_bf16(a0, bf, acc[0][j]);
        acc[1][j] = wmma_bf16(a1, bf, acc[1][j]);
      }
    }
  }

  // epilogue: bias + relu, fold gate into H rows, store bf16
  const int col = lane & 15, rowoff = (lane >> 4) * 8;
#pragma unroll
  for (int i = 0; i < 2; ++i) {
    float gv[8];
#pragma unroll
    for (int r = 0; r < 8; ++r)
      gv[r] = gate[(size_t)(m0 + m0w + i * 16 + rowoff + r) * NEXP + e];
#pragma unroll
    for (int j = 0; j < 4; ++j) {
      const int n = n0 + n0w + j * 16 + col;
      const float bb = b1e[n];
#pragma unroll
      for (int r = 0; r < 8; ++r) {
        float h = acc[i][j][r] + bb;
        h = h > 0.f ? h : 0.f;
        const size_t row = (size_t)(m0 + m0w + i * 16 + rowoff + r);
        Hb[row * DHID + n] = (__bf16)(h * gv[r]);
      }
    }
  }
}

// ---------------------------------------------------------------------------
// GEMM2 (per expert): out[m,n] (+)= Hb[m,:] @ W2b[e][:,n] + gate[m,e]*b2[e,n]
// ---------------------------------------------------------------------------
__global__ void __launch_bounds__(256)
gemm2_kernel(const unsigned short* __restrict__ Hb,
             const unsigned short* __restrict__ W2b,
             const float* __restrict__ b2, const float* __restrict__ gate,
             float* __restrict__ out, int e, int first) {
  __shared__ __align__(16) unsigned short Asm[2][128 * 64];

  const int tid  = threadIdx.x;
  const int lane = tid & 31;
  const int wave = tid >> 5;
  const int m0   = blockIdx.x * 128;
  const int n0   = blockIdx.y * 128;
  const int m0w  = (wave & 3) * 32;
  const int n0w  = (wave >> 2) * 64;

  const unsigned short* Agl = Hb + (size_t)m0 * DHID;
  const unsigned short* Bp  = W2b + (size_t)e * DHID * DOUT;  // panel layout
  const float* b2e = b2 + (size_t)e * DOUT;

  const unsigned lds0 = (unsigned)(unsigned long long)(&Asm[0][0]);
  const unsigned lds1 = (unsigned)(unsigned long long)(&Asm[1][0]);

  v8f acc[2][4];
#pragma unroll
  for (int i = 0; i < 2; ++i)
#pragma unroll
    for (int j = 0; j < 4; ++j) acc[i][j] = zero8();

  const int NS = DHID / 64;
  stage_async(Agl, DHID, 0, lds0, tid);
  for (int ks = 0; ks < NS; ++ks) {
    wait_async_all();
    __syncthreads();
    if (ks + 1 < NS)
      stage_async(Agl, DHID, (ks + 1) * 64, ((ks + 1) & 1) ? lds1 : lds0, tid);

    const unsigned short* S = &Asm[ks & 1][0];
#pragma unroll
    for (int ksub = 0; ksub < 2; ++ksub) {
      const v16bf a0 = load_a_lds64(S, m0w,      ksub, lane);
      const v16bf a1 = load_a_lds64(S, m0w + 16, ksub, lane);
      const int k0 = ks * 64 + ksub * 32;
#pragma unroll
      for (int j = 0; j < 4; ++j) {
        const v16bf bf = load_b_panel(Bp, DOUT, k0, n0 + n0w + j * 16, lane);
        acc[0][j] = wmma_bf16(a0, bf, acc[0][j]);
        acc[1][j] = wmma_bf16(a1, bf, acc[1][j]);
      }
    }
  }

  // epilogue: add gated bias, accumulate into out (expert 0 initializes;
  // block exclusively owns its out tile -> deterministic plain RMW)
  const int col = lane & 15, rowoff = (lane >> 4) * 8;
#pragma unroll
  for (int i = 0; i < 2; ++i) {
    float gv[8];
#pragma unroll
    for (int r = 0; r < 8; ++r)
      gv[r] = gate[(size_t)(m0 + m0w + i * 16 + rowoff + r) * NEXP + e];
#pragma unroll
    for (int j = 0; j < 4; ++j) {
      const int n = n0 + n0w + j * 16 + col;
      const float bb = b2e[n];
#pragma unroll
      for (int r = 0; r < 8; ++r) {
        const size_t row = (size_t)(m0 + m0w + i * 16 + rowoff + r);
        const float c = acc[i][j][r] + gv[r] * bb;
        if (first) out[row * DOUT + n] = c;
        else       out[row * DOUT + n] += c;
      }
    }
  }
}

// ---------------------------------------------------------------------------
extern "C" void kernel_launch(void* const* d_in, const int* in_sizes, int n_in,
                              void* d_out, int out_size, void* d_ws, size_t ws_size,
                              hipStream_t stream) {
  const float* x  = (const float*)d_in[0];
  const float* W1 = (const float*)d_in[1];
  const float* b1 = (const float*)d_in[2];
  const float* W2 = (const float*)d_in[3];
  const float* b2 = (const float*)d_in[4];
  const float* Wg = (const float*)d_in[5];
  const float* bg = (const float*)d_in[6];
  float* out      = (float*)d_out;

  // Workspace layout (bytes), all 256B aligned. Requires ws_size >= ~209MB.
  unsigned char* ws = (unsigned char*)d_ws;
  float*          gateBuf = (float*)(ws + 0);                       // 256KB
  __bf16*         xb      = (__bf16*)(ws + (size_t)262144);         // 16MB
  __bf16*         W1b     = (__bf16*)(ws + (size_t)17039360);       // 64MB
  __bf16*         W2b     = (__bf16*)(ws + (size_t)84148224);       // 64MB
  __bf16*         Hb      = (__bf16*)(ws + (size_t)151257088);      // 64MB (per-expert reuse)

  // One-time precision/layout prep (per launch): fp32 -> bf16, panelized W.
  cvt_x_kernel<<<(Bsz * DIN) / (256 * 4), 256, 0, stream>>>(x, xb);
  cvt_w_kernel<DIN,  DHID><<<(NEXP * DIN * DHID) / (256 * 4), 256, 0, stream>>>(W1, W1b);
  cvt_w_kernel<DHID, DOUT><<<(NEXP * DHID * DOUT) / (256 * 4), 256, 0, stream>>>(W2, W2b);

  // Gating softmax.
  moe_gate<<<Bsz / 8, 256, 0, stream>>>(x, Wg, bg, gateBuf);

  // Per expert: GEMM1 (gate folded into H) then GEMM2 accumulation.
  // Hb (64MB) is produced then immediately consumed -> stays in 192MB L2.
  for (int e = 0; e < NEXP; ++e) {
    gemm1_kernel<<<dim3(Bsz / 128, DHID / 128), 256, 0, stream>>>(
        (const unsigned short*)xb, (const unsigned short*)W1b, b1, gateBuf, Hb, e);
    gemm2_kernel<<<dim3(Bsz / 128, DOUT / 128), 256, 0, stream>>>(
        (const unsigned short*)Hb, (const unsigned short*)W2b, b2, gateBuf, out,
        e, e == 0 ? 1 : 0);
  }
}
